// TransformedAttention_89180700934767
// MI455X (gfx1250) — matrix operand order repaired and checked by reading, not compiled, not run on hardware
//
#include <hip/hip_runtime.h>
#include <hip/hip_bf16.h>

typedef _Float16 h16;
typedef __attribute__((ext_vector_type(16))) _Float16 v16h;
typedef __attribute__((ext_vector_type(8)))  _Float16 v8h;
typedef __attribute__((ext_vector_type(8)))  float    v8f;
typedef __attribute__((ext_vector_type(4)))  float    v4f;

#define B_  8
#define S_  2048
#define D_  512
#define U_  512
#define L0_ 128
#define L1_ 128

#define QS_STRIDE 520                 // 512 + 8 halves pad: 260 dwords = 4 mod 64 banks
#define QS_BYTES  (4 * 16 * QS_STRIDE * 2)   // 4 query tiles staged per block
#define P_BYTES   (8 * 16 * 32 * 2)          // per-wave 16x32 probability tile
#define SMEM_BYTES (QS_BYTES + P_BYTES)

__device__ __forceinline__ v8f wmma16(const v16h& a, const v16h& b, const v8f& c) {
  return __builtin_amdgcn_wmma_f32_16x16x32_f16(false, a, false, b, (short)0, c, false, false);
}

// ---------------- Kernel 1: W (DxU fp32, row-major) -> Wt (3 x U x D f16, transposed) ----
__global__ void wt_kernel(const float* __restrict__ Wq,
                          const float* __restrict__ Wk,
                          const float* __restrict__ Wv,
                          h16* __restrict__ Wt) {
  int tid = blockIdx.x * 256 + threadIdx.x;    // 3*U*D threads; Wt flat [w][n][k]
  int k = tid & (D_ - 1);
  int n = (tid >> 9) & (U_ - 1);
  int w = tid >> 18;
  const float* W = (w == 0) ? Wq : (w == 1) ? Wk : Wv;
  Wt[tid] = (h16)W[k * U_ + n];
}

// ---------------- Kernel 2: Q/K/V projections via WMMA ------------------------------
// grid (S/128, U/64, B*3), block 256. Wave w: rows [m0+16w, +16), cols [n0, n0+64).
// widx 0 -> qh row-major (scaled by 1/sqrt(U)), 1 -> kh row-major, 2 -> vth transposed [U][S].
__global__ __launch_bounds__(256) void proj_kernel(const float* __restrict__ x,
                                                   const h16* __restrict__ Wt,
                                                   const float* __restrict__ bq,
                                                   const float* __restrict__ bk,
                                                   const float* __restrict__ bv,
                                                   h16* __restrict__ qh,
                                                   h16* __restrict__ kh,
                                                   h16* __restrict__ vth) {
  const int lane = threadIdx.x & 31;
  const int wv   = threadIdx.x >> 5;
  const int r    = lane & 15;
  const int hi   = lane >> 4;
  const int m0   = blockIdx.x * 128 + wv * 16;
  const int n0   = blockIdx.y * 64;
  const int b    = blockIdx.z / 3;
  const int widx = blockIdx.z % 3;

  const float* xrow  = x + ((size_t)(b * S_) + m0 + r) * D_;
  const h16*   wbase = Wt + (size_t)widx * U_ * D_;

  v8f c[4];
#pragma unroll
  for (int f = 0; f < 4; ++f)
#pragma unroll
    for (int i = 0; i < 8; ++i) c[f][i] = 0.0f;

  for (int k0 = 0; k0 < D_; k0 += 32) {
    // A fragment (fp32 -> f16 on the fly); lane<16: K 0-15; lane>=16: K 16-31
    const float* p = xrow + k0 + hi * 16;
    v4f f0 = *(const v4f*)(p);
    v4f f1 = *(const v4f*)(p + 4);
    v4f f2 = *(const v4f*)(p + 8);
    v4f f3 = *(const v4f*)(p + 12);
    v16h a;
#pragma unroll
    for (int i = 0; i < 4; ++i) {
      a[i]      = (h16)f0[i];
      a[4 + i]  = (h16)f1[i];
      a[8 + i]  = (h16)f2[i];
      a[12 + i] = (h16)f3[i];
    }
#pragma unroll
    for (int f = 0; f < 4; ++f) {
      int n = n0 + f * 16 + r;   // B fragment: lane = output column, contiguous K halves
      v16h bf = *(const v16h*)(wbase + (size_t)n * D_ + k0 + hi * 16);
      c[f] = wmma16(a, bf, c[f]);
    }
  }

  const float* bias = (widx == 0) ? bq : (widx == 1) ? bk : bv;
  const float qscale = 0.04419417382415922f;  // 1/sqrt(512)

  if (widx < 2) {
    h16* out = (widx == 0) ? qh : kh;
#pragma unroll
    for (int f = 0; f < 4; ++f) {
      int col = n0 + f * 16 + r;
      float bb = bias[col];
#pragma unroll
      for (int g = 0; g < 8; ++g) {
        float v = c[f][g] + bb;
        if (widx == 0) v *= qscale;
        out[((size_t)(b * S_) + m0 + g + 8 * hi) * U_ + col] = (h16)v;
      }
    }
  } else {
#pragma unroll
    for (int f = 0; f < 4; ++f) {
      int col = n0 + f * 16 + r;
      float bb = bias[col];
      v8h vv;
#pragma unroll
      for (int g = 0; g < 8; ++g) vv[g] = (h16)(c[f][g] + bb);
      // 8 consecutive rows (g) of this column land contiguous in Vt[col][*]
      *(v8h*)(vth + ((size_t)(b * U_) + col) * S_ + m0 + 8 * hi) = vv;
    }
  }
}

// ---------------- Kernel 3: sliding-window flash attention via WMMA -----------------
// grid (S/64, B), block 256, dynamic LDS. Wave = (query tile qt 0..3) x (column half ch 0..1).
// Q tiles staged once in LDS (padded stride -> conflict-free ds_load_b128 A-fragments).
__global__ __launch_bounds__(256) void attn_kernel(const h16* __restrict__ qh,
                                                   const h16* __restrict__ kh,
                                                   const h16* __restrict__ vth,
                                                   float* __restrict__ out) {
  extern __shared__ __align__(32) char smem[];
  h16* Qs_all = (h16*)smem;                       // [4][16][QS_STRIDE]
  h16* P_all  = (h16*)(smem + QS_BYTES);          // [8][16][32]

  const int lane  = threadIdx.x & 31;
  const int wv    = threadIdx.x >> 5;
  const int qt    = wv >> 1;
  const int ch    = wv & 1;
  const int r     = lane & 15;
  const int hi    = lane >> 4;
  const int b     = blockIdx.y;
  const int qbase = blockIdx.x * 64 + qt * 16;
  const int n0    = ch * 256;

  h16* Qs = Qs_all + qt * 16 * QS_STRIDE;
  h16* Pw = P_all + wv * 16 * 32;

  const h16* kbatch = kh + (size_t)b * S_ * U_;
  const h16* vbatch = vth + (size_t)b * U_ * S_;

  // ---- stage this query tile's 16x512 f16 block into LDS (2 waves cooperate) ----
  {
    const int tt  = ch * 32 + lane;   // 0..63 within the qtile pair
    const int row = tt >> 2;          // 0..15
    const int seg = tt & 3;           // 4 x 128-half segments per row
    const h16* src = qh + ((size_t)(b * S_) + qbase + row) * U_ + seg * 128;
    h16* dst = Qs + row * QS_STRIDE + seg * 128;
#pragma unroll
    for (int i = 0; i < 16; ++i)
      *(v8h*)(dst + i * 8) = *(const v8h*)(src + i * 8);
  }
  __syncthreads();   // uniform: before any wave-divergent trip counts

  v8f acc[16];
#pragma unroll
  for (int f = 0; f < 16; ++f)
#pragma unroll
    for (int i = 0; i < 8; ++i) acc[f][i] = 0.0f;

  float mrow[8], lrow[8];
#pragma unroll
  for (int g = 0; g < 8; ++g) { mrow[g] = -1e30f; lrow[g] = 0.0f; }

  int kb_lo = qbase - L0_;
  if (kb_lo < 0) kb_lo = 0;                       // stays 16-aligned
  int kb_hi = qbase + 15 + L1_;
  if (kb_hi > S_ - 1) kb_hi = S_ - 1;

  const h16* qlds = Qs + r * QS_STRIDE;           // A-fragment row for this lane

  for (int kb = kb_lo; kb <= kb_hi; kb += 32) {
    const bool v1  = (kb + 16 <= kb_hi);          // wave-uniform validity of 2nd subtile
    int kt1 = kb + 16; if (kt1 > S_ - 16) kt1 = S_ - 16;  // clamped (masked if invalid)

    const h16* krow0 = kbatch + (size_t)(kb  + r) * U_;
    const h16* krow1 = kbatch + (size_t)(kt1 + r) * U_;
    if (kb + 32 <= kb_hi)
      __builtin_prefetch((const void*)(krow0 + 32 * (size_t)U_), 0, 1);

    v8f s0, s1;
#pragma unroll
    for (int i = 0; i < 8; ++i) { s0[i] = 0.0f; s1[i] = 0.0f; }

#pragma unroll 4
    for (int u = 0; u < U_; u += 32) {
      v8h alo = *(const v8h*)(qlds + u + hi * 16);
      v8h ahi = *(const v8h*)(qlds + u + hi * 16 + 8);
      v16h a;
#pragma unroll
      for (int i = 0; i < 8; ++i) { a[i] = alo[i]; a[8 + i] = ahi[i]; }
      v16h b0 = *(const v16h*)(krow0 + u + hi * 16);
      v16h b1 = *(const v16h*)(krow1 + u + hi * 16);
      s0 = wmma16(a, b0, s0);
      s1 = wmma16(a, b1, s1);
    }

    // joint online softmax over the 32-key pair; row M = g + 8*hi lives in one 16-lane half
    const int j0 = kb + r;
    const int j1 = kb + 16 + r;
#pragma unroll
    for (int g = 0; g < 8; ++g) {
      int q = qbase + g + 8 * hi;
      float t0 = s0[g], t1 = s1[g];
      if (j0 < q - L0_ || j0 > q + L1_) t0 = -3.0e38f;
      if (!v1 || j1 < q - L0_ || j1 > q + L1_) t1 = -3.0e38f;
      float rmax = fmaxf(t0, t1);
      rmax = fmaxf(rmax, __shfl_xor(rmax, 1));
      rmax = fmaxf(rmax, __shfl_xor(rmax, 2));
      rmax = fmaxf(rmax, __shfl_xor(rmax, 4));
      rmax = fmaxf(rmax, __shfl_xor(rmax, 8));
      float newm  = fmaxf(mrow[g], rmax);
      float scale = __expf(mrow[g] - newm);
      float e0 = __expf(t0 - newm);
      float e1 = __expf(t1 - newm);
      float rs = e0 + e1;
      rs += __shfl_xor(rs, 1);
      rs += __shfl_xor(rs, 2);
      rs += __shfl_xor(rs, 4);
      rs += __shfl_xor(rs, 8);
      lrow[g] = lrow[g] * scale + rs;
      mrow[g] = newm;
#pragma unroll
      for (int f = 0; f < 16; ++f) acc[f][g] *= scale;
      Pw[(g + 8 * hi) * 32 + r]      = (h16)e0;   // C-layout -> LDS
      Pw[(g + 8 * hi) * 32 + 16 + r] = (h16)e1;
    }
    asm volatile("s_wait_dscnt 0" ::: "memory");  // same-wave DS in-order; re-read as A-layout

    v8h plo = *(const v8h*)(Pw + r * 32 + hi * 16);
    v8h phi = *(const v8h*)(Pw + r * 32 + hi * 16 + 8);
    v16h pa;
#pragma unroll
    for (int i = 0; i < 8; ++i) { pa[i] = plo[i]; pa[8 + i] = phi[i]; }

    int kr = kb + hi * 16;
    if (kr > S_ - 16) kr = S_ - 16;               // clamped rows carry zero probability
#pragma unroll
    for (int f = 0; f < 16; ++f) {
      int col = n0 + f * 16 + r;
      v16h bf = *(const v16h*)(vbatch + (size_t)col * S_ + kr);
      acc[f] = wmma16(pa, bf, acc[f]);
    }
  }

#pragma unroll
  for (int g = 0; g < 8; ++g) {
    int q = qbase + g + 8 * hi;
    float inv = 1.0f / lrow[g];
    float* orow = out + ((size_t)(b * S_) + q) * U_;
#pragma unroll
    for (int f = 0; f < 16; ++f) orow[n0 + f * 16 + r] = acc[f][g] * inv;
  }
}

extern "C" void kernel_launch(void* const* d_in, const int* in_sizes, int n_in,
                              void* d_out, int out_size, void* d_ws, size_t ws_size,
                              hipStream_t stream) {
  const float* x  = (const float*)d_in[0];
  const float* Wq = (const float*)d_in[1];
  const float* Wk = (const float*)d_in[2];
  const float* Wv = (const float*)d_in[3];
  const float* bq = (const float*)d_in[4];
  const float* bk = (const float*)d_in[5];
  const float* bv = (const float*)d_in[6];
  float* out = (float*)d_out;

  char* ws = (char*)d_ws;
  h16* Wt  = (h16*)ws;                                  // 3*U*D halves (1.5 MB)
  h16* qh  = (h16*)(ws + (size_t)3 * U_ * D_ * 2);      // B*S*U halves (16 MB)
  h16* kh  = qh + (size_t)B_ * S_ * U_;                 // 16 MB
  h16* vth = kh + (size_t)B_ * S_ * U_;                 // 16 MB, transposed [B][U][S]

  wt_kernel<<<(3 * U_ * D_) / 256, 256, 0, stream>>>(Wq, Wk, Wv, Wt);
  proj_kernel<<<dim3(S_ / 128, U_ / 64, B_ * 3), 256, 0, stream>>>(x, Wt, bq, bk, bv, qh, kh, vth);
  attn_kernel<<<dim3(S_ / 64, B_), 256, SMEM_BYTES, stream>>>(qh, kh, vth, out);
}
